// CANN_60455959658874
// MI455X (gfx1250) — matrix operand order repaired and checked by reading, not compiled, not run on hardware
//
#include <hip/hip_runtime.h>
#include <hip/hip_bf16.h>

typedef __attribute__((ext_vector_type(16))) __bf16 v16bf;
typedef __attribute__((ext_vector_type(8)))  __bf16 v8bf;
typedef __attribute__((ext_vector_type(8)))  float  v8f;

#define DIM 2048
constexpr int BM = 128;
constexpr int BN = 128;
constexpr int BK = 64;
constexpr int LDS_K = BK + 8;   // 72 halves = 144B row stride (16B multiple, bank-skewed)

// ---------------------------------------------------------------------------
// f32 -> bf16 conversion (4 elements / thread)
// ---------------------------------------------------------------------------
__global__ __launch_bounds__(256)
void cast_bf16_kernel(const float* __restrict__ in, __bf16* __restrict__ out, int n) {
    int i = (blockIdx.x * 256 + threadIdx.x) * 4;
    if (i + 3 < n) {
        float4 f = *(const float4*)(in + i);
        out[i + 0] = (__bf16)f.x;
        out[i + 1] = (__bf16)f.y;
        out[i + 2] = (__bf16)f.z;
        out[i + 3] = (__bf16)f.w;
    }
}

// ---------------------------------------------------------------------------
// Canonical GEMM:  C(MxN) = A(MxK) * B(NxK)^T  (+ bias[n])
// OUT_MODE: 0 = f32 row-major, 1 = bf16 row-major, 2 = bf16 transposed (N x M)
// Double-buffered LDS pipeline: prefetch tile kb+1 from global into registers
// while computing tile kb from LDS, then store into the alternate LDS stage.
// ---------------------------------------------------------------------------
template <int OUT_MODE, bool HAS_BIAS>
__global__ __launch_bounds__(256)
void gemm_bt_kernel(const __bf16* __restrict__ A, const __bf16* __restrict__ B,
                    const float* __restrict__ bias, void* __restrict__ Cout,
                    int M, int N, int K)
{
    __shared__ __attribute__((aligned(16))) __bf16 As[2][BM * LDS_K];
    __shared__ __attribute__((aligned(16))) __bf16 Bs[2][BN * LDS_K];

    const int tid    = threadIdx.x;
    const int lane   = tid & 31;
    const int wave   = tid >> 5;          // 0..7
    const int wm     = wave >> 2;         // 0..1 -> 64-row strip
    const int wn     = wave & 3;          // 0..3 -> 32-col strip
    const int blockM = blockIdx.y * BM;
    const int blockN = blockIdx.x * BN;

    v8f acc[4][2];
    #pragma unroll
    for (int i = 0; i < 4; ++i)
        #pragma unroll
        for (int j = 0; j < 2; ++j)
            acc[i][j] = (v8f){0.f, 0.f, 0.f, 0.f, 0.f, 0.f, 0.f, 0.f};

    // Global->LDS mapping: 256 threads x 16B, 4 row-groups of 32 rows each.
    const int lrow = tid >> 3;            // 0..31
    const int lcol = (tid & 7) * 8;       // 0..56 (halves)

    const int mrow = lane & 15;           // frag row (A) / column (B, C)
    const int hi   = lane >> 4;           // lane-half select

    const __bf16* gA = A + (size_t)(blockM + lrow) * K + lcol;
    const __bf16* gB = B + (size_t)(blockN + lrow) * K + lcol;
    const size_t gRowStep = (size_t)32 * K;   // 32-row group stride

    v8bf pa[4], pb[4];

    // ---- prologue: fetch tile 0 into stage 0 ----
    #pragma unroll
    for (int i = 0; i < 4; ++i) {
        pa[i] = *(const v8bf*)(gA + i * gRowStep);
        pb[i] = *(const v8bf*)(gB + i * gRowStep);
    }
    #pragma unroll
    for (int i = 0; i < 4; ++i) {
        *(v8bf*)(As[0] + (lrow + 32 * i) * LDS_K + lcol) = pa[i];
        *(v8bf*)(Bs[0] + (lrow + 32 * i) * LDS_K + lcol) = pb[i];
    }
    __syncthreads();

    auto compute_tile = [&](const __bf16* AsP, const __bf16* BsP) {
        #pragma unroll
        for (int ks = 0; ks < BK; ks += 32) {
            // A fragments (ISA 7.12.2: lane half picks K {0-7,16-23} / {8-15,24-31})
            v16bf afr[4];
            #pragma unroll
            for (int i = 0; i < 4; ++i) {
                const __bf16* p = AsP + (wm * 64 + i * 16 + mrow) * LDS_K + ks + hi * 8;
                v8bf lo  = *(const v8bf*)(p);
                v8bf hi8 = *(const v8bf*)(p + 16);
                afr[i] = __builtin_shufflevector(lo, hi8,
                         0,1,2,3,4,5,6,7,8,9,10,11,12,13,14,15);
            }
            // B fragments: lane = column, same K split as A ({0-7,16-23} / {8-15,24-31})
            v16bf bfr[2];
            #pragma unroll
            for (int j = 0; j < 2; ++j) {
                const __bf16* p = BsP + (wn * 32 + j * 16 + mrow) * LDS_K + ks + hi * 8;
                v8bf lo  = *(const v8bf*)(p);
                v8bf hi8 = *(const v8bf*)(p + 16);
                bfr[j] = __builtin_shufflevector(lo, hi8,
                         0,1,2,3,4,5,6,7,8,9,10,11,12,13,14,15);
            }
            #pragma unroll
            for (int i = 0; i < 4; ++i)
                #pragma unroll
                for (int j = 0; j < 2; ++j)
                    acc[i][j] = __builtin_amdgcn_wmma_f32_16x16x32_bf16(
                        false, afr[i], false, bfr[j], (short)0, acc[i][j],
                        false, false);
        }
    };

    const int nb = K / BK;
    // ---- steady state: prefetch kb+1 while computing kb ----
    for (int kb = 0; kb < nb - 1; ++kb) {
        const int cur = kb & 1;
        const int nxt = cur ^ 1;
        const int k1  = (kb + 1) * BK;
        #pragma unroll
        for (int i = 0; i < 4; ++i) {
            pa[i] = *(const v8bf*)(gA + k1 + i * gRowStep);
            pb[i] = *(const v8bf*)(gB + k1 + i * gRowStep);
        }
        compute_tile(As[cur], Bs[cur]);
        #pragma unroll
        for (int i = 0; i < 4; ++i) {
            *(v8bf*)(As[nxt] + (lrow + 32 * i) * LDS_K + lcol) = pa[i];
            *(v8bf*)(Bs[nxt] + (lrow + 32 * i) * LDS_K + lcol) = pb[i];
        }
        __syncthreads();
    }
    // ---- last tile ----
    compute_tile(As[(nb - 1) & 1], Bs[(nb - 1) & 1]);

    // Epilogue. C/D layout: VGPR r -> M = r + 8*(lane>=16), lane&15 -> N.
    const int moff = hi * 8;
    #pragma unroll
    for (int i = 0; i < 4; ++i) {
        #pragma unroll
        for (int j = 0; j < 2; ++j) {
            const int mBase = blockM + wm * 64 + i * 16 + moff;
            const int nG    = blockN + wn * 32 + j * 16 + mrow;
            const float bv  = HAS_BIAS ? bias[nG] : 0.0f;
            if (OUT_MODE == 2) {
                // transposed store: 8 consecutive M per lane -> contiguous 16B
                v8bf pack;
                #pragma unroll
                for (int r = 0; r < 8; ++r)
                    pack[r] = (__bf16)(acc[i][j][r] + bv);
                *(v8bf*)((__bf16*)Cout + (size_t)nG * M + mBase) = pack;
            } else {
                #pragma unroll
                for (int r = 0; r < 8; ++r) {
                    const float v = acc[i][j][r] + bv;
                    const size_t idx = (size_t)(mBase + r) * N + nG;
                    if (OUT_MODE == 0) ((float*)Cout)[idx] = v;
                    else               ((__bf16*)Cout)[idx] = (__bf16)v;
                }
            }
        }
    }
}

// ---------------------------------------------------------------------------
// Row softmax of scores*SCALE -> bf16 attn. One 256-thread block per row.
// ---------------------------------------------------------------------------
__global__ __launch_bounds__(256)
void softmax_kernel(const float* __restrict__ S, __bf16* __restrict__ attn, int N)
{
    __shared__ float red[256];
    const int tid = threadIdx.x;
    const float* s = S + (size_t)blockIdx.x * N;

    float vals[8];
    float lmax = -3.4e38f;
    #pragma unroll
    for (int i = 0; i < 8; ++i) {
        vals[i] = s[tid + i * 256] * 0.125f;   // SCALE = 64^-0.5
        lmax = fmaxf(lmax, vals[i]);
    }
    red[tid] = lmax;
    __syncthreads();
    for (int off = 128; off > 0; off >>= 1) {
        if (tid < off) red[tid] = fmaxf(red[tid], red[tid + off]);
        __syncthreads();
    }
    const float m = red[0];
    __syncthreads();

    float lsum = 0.f;
    #pragma unroll
    for (int i = 0; i < 8; ++i) {
        vals[i] = __expf(vals[i] - m);
        lsum += vals[i];
    }
    red[tid] = lsum;
    __syncthreads();
    for (int off = 128; off > 0; off >>= 1) {
        if (tid < off) red[tid] += red[tid + off];
        __syncthreads();
    }
    const float inv = 1.0f / red[0];
    __bf16* a = attn + (size_t)blockIdx.x * N;
    #pragma unroll
    for (int i = 0; i < 8; ++i)
        a[tid + i * 256] = (__bf16)(vals[i] * inv);
}

// ---------------------------------------------------------------------------
extern "C" void kernel_launch(void* const* d_in, const int* in_sizes, int n_in,
                              void* d_out, int out_size, void* d_ws, size_t ws_size,
                              hipStream_t stream)
{
    const float* z  = (const float*)d_in[0];
    const float* Wq = (const float*)d_in[1];
    const float* bq = (const float*)d_in[2];
    const float* Wk = (const float*)d_in[3];
    const float* bk = (const float*)d_in[4];
    const float* Wv = (const float*)d_in[5];
    const float* bv = (const float*)d_in[6];
    float* out = (float*)d_out;

    char* ws = (char*)d_ws;
    __bf16* Abf = (__bf16*)(ws);                       //  8 MB  z (flattened) bf16
    __bf16* Qb  = (__bf16*)(ws + (8ull  << 20));       //  8 MB
    __bf16* Kb  = (__bf16*)(ws + (16ull << 20));       //  8 MB
    __bf16* VTb = (__bf16*)(ws + (24ull << 20));       //  8 MB  V transposed
    __bf16* Wb  = (__bf16*)(ws + (32ull << 20));       //  8 MB  reused per-projection
    float*  Sf  = (float*) (ws + (40ull << 20));       // 16 MB  scores f32
    __bf16* ATb = (__bf16*)(ws + (56ull << 20));       //  8 MB  attn bf16  (total 64 MB)

    const int NELT = DIM * DIM;
    const dim3 cgrid(NELT / (256 * 4)), cblk(256);
    const dim3 ggrid(DIM / BN, DIM / BM), gblk(256);

    // z -> bf16
    cast_bf16_kernel<<<cgrid, cblk, 0, stream>>>(z, Abf, NELT);

    // Q = zf Wq^T + bq   (bf16 out)
    cast_bf16_kernel<<<cgrid, cblk, 0, stream>>>(Wq, Wb, NELT);
    gemm_bt_kernel<1, true><<<ggrid, gblk, 0, stream>>>(Abf, Wb, bq, Qb, DIM, DIM, DIM);

    // K = zf Wk^T + bk   (bf16 out)
    cast_bf16_kernel<<<cgrid, cblk, 0, stream>>>(Wk, Wb, NELT);
    gemm_bt_kernel<1, true><<<ggrid, gblk, 0, stream>>>(Abf, Wb, bk, Kb, DIM, DIM, DIM);

    // V = zf Wv^T + bv   (bf16, stored transposed so attn@V is also A*B^T)
    cast_bf16_kernel<<<cgrid, cblk, 0, stream>>>(Wv, Wb, NELT);
    gemm_bt_kernel<2, true><<<ggrid, gblk, 0, stream>>>(Abf, Wb, bv, VTb, DIM, DIM, DIM);

    // scores = Q K^T   (f32 out; scale folded into softmax)
    gemm_bt_kernel<0, false><<<ggrid, gblk, 0, stream>>>(Qb, Kb, nullptr, Sf, DIM, DIM, DIM);

    // attn = softmax(scores * 0.125)  -> bf16
    softmax_kernel<<<dim3(DIM), dim3(256), 0, stream>>>(Sf, ATb, DIM);

    // h = attn V  =  attn * (V^T)^T   (f32 out to d_out)
    gemm_bt_kernel<0, false><<<ggrid, gblk, 0, stream>>>(ATb, VTb, nullptr, out, DIM, DIM, DIM);
}
